// AppearancePoolFusion_52948356825728
// MI455X (gfx1250) — matrix operand order repaired and checked by reading, not compiled, not run on hardware
//
#include <hip/hip_runtime.h>
#include <hip/hip_bf16.h>

#define N_NODES 8192
#define N_EDGES 131072
#define NF 128
#define LRELU_S 0.2f

typedef __attribute__((ext_vector_type(16))) __bf16 v16bf;
typedef __attribute__((ext_vector_type(8)))  float  v8f;
typedef __attribute__((ext_vector_type(4)))  unsigned int v4u;

union FragBF { v16bf bf; v4u q[2]; };

__device__ __forceinline__ unsigned short f2bf(float f) {
  unsigned int u = __float_as_uint(f);
  unsigned int r = u + 0x7FFFu + ((u >> 16) & 1u);
  return (unsigned short)(r >> 16);
}

__device__ __forceinline__ void atomicMaxF(float* addr, float val) {
  unsigned int* ua = (unsigned int*)addr;
  unsigned int old = *ua;
  while (__uint_as_float(old) < val) {
    unsigned int assumed = old;
    old = atomicCAS(ua, assumed, __float_as_uint(val));
    if (old == assumed) break;
  }
}

// ---------------- elementwise helpers ----------------

__global__ void k_fill_f32(float* __restrict__ p, float v, int n) {
  int i = blockIdx.x * blockDim.x + threadIdx.x;
  if (i < n) p[i] = v;
}

__global__ void k_pack_bf16(const float* __restrict__ a, unsigned short* __restrict__ o, int n) {
  int i = blockIdx.x * blockDim.x + threadIdx.x;
  if (i < n) o[i] = f2bf(a[i]);
}

// rows x (wa+wb): concat two fp32 matrices -> one bf16 matrix
__global__ void k_pack2_bf16(const float* __restrict__ a, int wa,
                             const float* __restrict__ b, int wb,
                             unsigned short* __restrict__ o, int rows) {
  int W = wa + wb;
  int i = blockIdx.x * blockDim.x + threadIdx.x;
  if (i >= rows * W) return;
  int r = i / W, c = i - r * W;
  float v = (c < wa) ? a[(size_t)r * wa + c] : b[(size_t)r * wb + (c - wa)];
  o[i] = f2bf(v);
}

// Pack fp32 W[K x NFo] (k-major) into per-wave B-fragment layout:
// Wp[((cb*(K/32)+s)*32 + lane)*16 + j] = bf16( W[s*32 + (lane>=16?8:0) + (j&7) + (j>=8?16:0)][cb*16 + lane%16] )
__global__ void k_pack_w(const float* __restrict__ Wm, unsigned short* __restrict__ Wp, int K, int NFo) {
  int t = blockIdx.x * blockDim.x + threadIdx.x;
  if (t >= K * NFo) return;
  int kb = K >> 5;
  int j = t & 15;
  int l = (t >> 4) & 31;
  int s = (t >> 9) % kb;
  int cb = t / (512 * kb);
  int k = s * 32 + ((l >= 16) ? 8 : 0) + (j & 7) + ((j >= 8) ? 16 : 0);
  int col = cb * 16 + (l & 15);
  Wp[t] = f2bf(Wm[(size_t)k * NFo + col]);
}

// ---------------- core WMMA edge-MLP ----------------
// Per-edge row input = [ xcat[src] (dx) | xcat[dst] (dx) | eA (wA) | eB (wB) ], K = 2*dx+wA+wB.
// Each wave: 16 edges x 64 output cols (4 WMMA accumulators sharing one A fragment).
// msg_out[r,c] = relu(row . W[:,c] + bias[c])  (bf16)
// node_acc[dst[r], c] += (pre-relu value) * emask[r]   (fp32 atomics)
__launch_bounds__(128)
__global__ void k_edge_mlp(const unsigned short* __restrict__ xcat, int dx,
                           const unsigned short* __restrict__ eA, int wA,
                           const unsigned short* __restrict__ eB, int wB,
                           const int* __restrict__ src, const int* __restrict__ dst,
                           const unsigned short* __restrict__ Wp,
                           const float* __restrict__ bias,
                           int K, int NFo,
                           unsigned short* __restrict__ msg_out,
                           float* __restrict__ node_acc,
                           const float* __restrict__ emask) {
  int lane = threadIdx.x & 31;
  int wave = blockIdx.x * (blockDim.x >> 5) + (threadIdx.x >> 5);
  int e0 = wave << 4;
  if (e0 >= N_EDGES) return;
  int cbase = blockIdx.y << 2;  // first 16-col tile of this wave's 64-col strip

  int arow = e0 + (lane & 15);
  int si = src[arow], di = dst[arow];
  int aoff = (lane < 16) ? 0 : 8;
  int kb = K >> 5;

  // segment table (branch-free hot loop: sequential sub-loops, uniform strides)
  const unsigned short* segp[4];
  int segn[4];
  segp[0] = xcat + (size_t)si * dx + aoff;   segn[0] = dx >> 5;
  segp[1] = xcat + (size_t)di * dx + aoff;   segn[1] = dx >> 5;
  segp[2] = eA + (size_t)arow * wA + aoff;   segn[2] = wA >> 5;
  segp[3] = (wB > 0) ? (eB + (size_t)arow * wB + aoff) : segp[2];
  segn[3] = wB >> 5;

  // four B-fragment streams, stride 512 ushorts per 32-K block
  const unsigned short* wp0 = Wp + ((size_t)(cbase + 0) * kb * 32 + lane) * 16;
  const unsigned short* wp1 = Wp + ((size_t)(cbase + 1) * kb * 32 + lane) * 16;
  const unsigned short* wp2 = Wp + ((size_t)(cbase + 2) * kb * 32 + lane) * 16;
  const unsigned short* wp3 = Wp + ((size_t)(cbase + 3) * kb * 32 + lane) * 16;

  v8f acc0 = {0.f, 0.f, 0.f, 0.f, 0.f, 0.f, 0.f, 0.f};
  v8f acc1 = acc0, acc2 = acc0, acc3 = acc0;

  for (int seg = 0; seg < 4; ++seg) {
    const unsigned short* bp = segp[seg];
    int nb = segn[seg];
    for (int s = 0; s < nb; ++s) {
      FragBF A;
      A.q[0] = *(const v4u*)(bp);
      A.q[1] = *(const v4u*)(bp + 16);
      bp += 32;
      FragBF B;
      B.q[0] = *(const v4u*)(wp0);
      B.q[1] = *(const v4u*)(wp0 + 8);
      acc0 = __builtin_amdgcn_wmma_f32_16x16x32_bf16(false, A.bf, false, B.bf,
                                                     (short)0, acc0, false, false);
      B.q[0] = *(const v4u*)(wp1);
      B.q[1] = *(const v4u*)(wp1 + 8);
      acc1 = __builtin_amdgcn_wmma_f32_16x16x32_bf16(false, A.bf, false, B.bf,
                                                     (short)0, acc1, false, false);
      B.q[0] = *(const v4u*)(wp2);
      B.q[1] = *(const v4u*)(wp2 + 8);
      acc2 = __builtin_amdgcn_wmma_f32_16x16x32_bf16(false, A.bf, false, B.bf,
                                                     (short)0, acc2, false, false);
      B.q[0] = *(const v4u*)(wp3);
      B.q[1] = *(const v4u*)(wp3 + 8);
      acc3 = __builtin_amdgcn_wmma_f32_16x16x32_bf16(false, A.bf, false, B.bf,
                                                     (short)0, acc3, false, false);
      wp0 += 512; wp1 += 512; wp2 += 512; wp3 += 512;
    }
  }

  // epilogue: bias + relu-store (bf16) + masked fp32 scatter to node accumulator
  int rbase = e0 + ((lane < 16) ? 0 : 8);
  int dstr[8];
  float emr[8];
#pragma unroll
  for (int i = 0; i < 8; ++i) {
    int r = rbase + i;
    dstr[i] = dst[r];
    emr[i] = emask ? emask[r] : 1.0f;
  }
  int col0 = (cbase << 4) + (lane & 15);
  v8f accs[4] = {acc0, acc1, acc2, acc3};
#pragma unroll
  for (int t = 0; t < 4; ++t) {
    int col = col0 + (t << 4);
    float bv = bias[col];
#pragma unroll
    for (int i = 0; i < 8; ++i) {
      int r = rbase + i;
      float v = accs[t][i] + bv;
      msg_out[(size_t)r * NFo + col] = f2bf(v > 0.f ? v : 0.f);
      if (emr[i] != 0.f)
        atomicAdd(node_acc + (size_t)dstr[i] * NFo + col, v * emr[i]);
    }
  }
}

__global__ void k_deg(const int* __restrict__ dst, const float* __restrict__ em,
                      float* __restrict__ deg) {
  int e = blockIdx.x * blockDim.x + threadIdx.x;
  if (e >= N_EDGES) return;
  float m = em ? em[e] : 1.f;
  if (m != 0.f) atomicAdd(deg + dst[e], m);
}

__global__ void k_node_fin(const float* __restrict__ acc, const float* __restrict__ deg,
                           float* __restrict__ out, int NFo) {
  int i = blockIdx.x * blockDim.x + threadIdx.x;
  if (i >= N_NODES * NFo) return;
  int r = i / NFo;
  float v = acc[i] / fmaxf(deg[r], 1.f);
  out[i] = v > 0.f ? v : 0.f;
}

// ---------------- GAT score + SAG pool ----------------

__global__ void k_dot_w(const float* __restrict__ x, const float* __restrict__ wg,
                        float* __restrict__ hs, int D) {
  int wv = (blockIdx.x * blockDim.x + threadIdx.x) >> 5;
  int lane = threadIdx.x & 31;
  if (wv >= N_NODES) return;
  float sum = 0.f;
  for (int k = lane; k < D; k += 32) sum += x[(size_t)wv * D + k] * wg[k];
  for (int o = 16; o > 0; o >>= 1) sum += __shfl_xor(sum, o, 32);
  if (lane == 0) hs[wv] = sum;
}

__global__ void k_gat_init(const float* __restrict__ hs, const float* __restrict__ ag,
                           float* __restrict__ mx, float* __restrict__ den,
                           float* __restrict__ num) {
  int n = blockIdx.x * blockDim.x + threadIdx.x;
  if (n >= N_NODES) return;
  float v = hs[n] * (ag[0] + ag[1]);
  v = v > 0.f ? v : LRELU_S * v;
  mx[n] = v;  // e_self; edge maxes merged on top
  den[n] = 0.f;
  num[n] = 0.f;
}

__global__ void k_gat_emax(const float* __restrict__ hs, const int* __restrict__ src,
                           const int* __restrict__ dst, const float* __restrict__ em,
                           const float* __restrict__ ag, float* __restrict__ mx) {
  int e = blockIdx.x * blockDim.x + threadIdx.x;
  if (e >= N_EDGES) return;
  float m = em ? em[e] : 1.f;
  if (m == 0.f) return;
  float v = hs[src[e]] * ag[0] + hs[dst[e]] * ag[1];
  v = v > 0.f ? v : LRELU_S * v;
  atomicMaxF(mx + dst[e], v);
}

__global__ void k_gat_esum(const float* __restrict__ hs, const int* __restrict__ src,
                           const int* __restrict__ dst, const float* __restrict__ em,
                           const float* __restrict__ ag, const float* __restrict__ mx,
                           float* __restrict__ den, float* __restrict__ num) {
  int e = blockIdx.x * blockDim.x + threadIdx.x;
  if (e >= N_EDGES) return;
  float m = em ? em[e] : 1.f;
  if (m == 0.f) return;
  int d = dst[e];
  float v = hs[src[e]] * ag[0] + hs[d] * ag[1];
  v = v > 0.f ? v : LRELU_S * v;
  float ex = expf(v - mx[d]);
  atomicAdd(den + d, ex);
  atomicAdd(num + d, ex * hs[src[e]]);
}

__global__ void k_gat_score(const float* __restrict__ hs, const float* __restrict__ ag,
                            const float* __restrict__ bg, const float* __restrict__ mx,
                            const float* __restrict__ den, const float* __restrict__ num,
                            float* __restrict__ score) {
  int n = blockIdx.x * blockDim.x + threadIdx.x;
  if (n >= N_NODES) return;
  float es = hs[n] * (ag[0] + ag[1]);
  es = es > 0.f ? es : LRELU_S * es;
  float exs = expf(es - mx[n]);
  score[n] = (num[n] + exs * hs[n]) / (den[n] + exs) + bg[0];
}

__global__ void k_ukey(const float* __restrict__ score, const float* __restrict__ nmask,
                       unsigned int* __restrict__ uk) {
  int n = blockIdx.x * blockDim.x + threadIdx.x;
  if (n >= N_NODES) return;
  unsigned int b = __float_as_uint(score[n]);
  unsigned int u = (b & 0x80000000u) ? ~b : (b | 0x80000000u);
  if (nmask && nmask[n] == 0.f) u = 0u;  // masked-out -> smallest key
  uk[n] = u;
}

// single-block radix select of k-th largest + index-ordered tie break -> keep mask
__global__ void k_topk(const unsigned int* __restrict__ uk, int k, float* __restrict__ keep) {
  __shared__ unsigned int hist[256];
  __shared__ unsigned int sPref;
  __shared__ int sKK;
  __shared__ unsigned int sflag[256];
  __shared__ unsigned int runBase;
  int tid = threadIdx.x;
  if (tid == 0) { sPref = 0u; sKK = k; }
  __syncthreads();
  for (int pass = 3; pass >= 0; --pass) {
    int shift = pass * 8;
    unsigned int hiMask = (pass == 3) ? 0u : (0xFFFFFFFFu << (shift + 8));
    hist[tid] = 0u;
    __syncthreads();
    unsigned int pref = sPref;
    for (int i = tid; i < N_NODES; i += 256) {
      unsigned int u = uk[i];
      if ((u & hiMask) == (pref & hiMask))
        atomicAdd(&hist[(u >> shift) & 255], 1u);
    }
    __syncthreads();
    if (tid == 0) {
      unsigned int cum = 0;
      int kk = sKK;
      for (int b = 255; b >= 0; --b) {
        unsigned int c = hist[b];
        if ((int)(cum + c) >= kk) {
          sPref = pref | ((unsigned int)b << shift);
          sKK = kk - (int)cum;
          break;
        }
        cum += c;
      }
    }
    __syncthreads();
  }
  unsigned int T = sPref;
  int ties = sKK;
  if (tid == 0) runBase = 0u;
  __syncthreads();
  for (int base = 0; base < N_NODES; base += 256) {
    int i = base + tid;
    unsigned int u = uk[i];
    unsigned int isEq = (u == T) ? 1u : 0u;
    sflag[tid] = isEq;
    __syncthreads();
    if (tid == 0) {
      unsigned int run = runBase;
      for (int j = 0; j < 256; ++j) { unsigned int t = sflag[j]; sflag[j] = run; run += t; }
      runBase = run;
    }
    __syncthreads();
    int kp = (u > T) || (isEq && (int)sflag[tid] < ties);
    keep[i] = kp ? 1.f : 0.f;
    __syncthreads();
  }
}

__global__ void k_pool_x(const float* __restrict__ x, const float* __restrict__ keep,
                         const float* __restrict__ score, float* __restrict__ out, int D) {
  int i = blockIdx.x * blockDim.x + threadIdx.x;
  if (i >= N_NODES * D) return;
  int r = i / D;
  out[i] = (keep[r] > 0.f) ? x[i] * tanhf(score[r]) : x[i];
}

__global__ void k_emask(const float* __restrict__ emPrev, const float* __restrict__ keep,
                        const int* __restrict__ src, const int* __restrict__ dst,
                        float* __restrict__ emNew) {
  int e = blockIdx.x * blockDim.x + threadIdx.x;
  if (e >= N_EDGES) return;
  float m = emPrev ? emPrev[e] : 1.f;
  emNew[e] = m * keep[src[e]] * keep[dst[e]];
}

__global__ void k_resid(const float* __restrict__ a, const float* __restrict__ b,
                        const float* __restrict__ mask, float* __restrict__ out, int D) {
  int i = blockIdx.x * blockDim.x + threadIdx.x;
  if (i >= N_NODES * D) return;
  int r = i / D;
  out[i] = (mask[r] > 0.f) ? a[i] + b[i] : a[i];
}

__global__ void k_final(const float* __restrict__ x, const float* __restrict__ wl,
                        const float* __restrict__ bl, float* __restrict__ out) {
  int n = blockIdx.x * blockDim.x + threadIdx.x;
  if (n >= N_NODES) return;
  float a0 = bl[0], a1 = bl[1], a2 = bl[2], a3 = bl[3];
  const float* xr = x + (size_t)n * NF;
  for (int k = 0; k < NF; ++k) {
    float v = xr[k];
    a0 += v * wl[k * 4 + 0];
    a1 += v * wl[k * 4 + 1];
    a2 += v * wl[k * 4 + 2];
    a3 += v * wl[k * 4 + 3];
  }
  float* o = out + (size_t)n * 4;
  o[0] = a0; o[1] = a1; o[2] = a2; o[3] = a3;
}

// ---------------- host orchestration ----------------

extern "C" void kernel_launch(void* const* d_in, const int* in_sizes, int n_in,
                              void* d_out, int out_size, void* d_ws, size_t ws_size,
                              hipStream_t stream) {
  (void)in_sizes; (void)n_in; (void)out_size; (void)ws_size;
  const int N = N_NODES, E = N_EDGES;

  const float* node_feat = (const float*)d_in[0];
  const int*   eidx      = (const int*)d_in[1];
  const float* edge_feat = (const float*)d_in[2];
  const float *w1 = (const float*)d_in[3],  *b1  = (const float*)d_in[4];
  const float *w2 = (const float*)d_in[5],  *b2  = (const float*)d_in[6];
  const float *w3 = (const float*)d_in[7],  *b3  = (const float*)d_in[8];
  const float *w3p = (const float*)d_in[9], *b3p = (const float*)d_in[10];
  const float *wg1 = (const float*)d_in[11], *ag1 = (const float*)d_in[12], *bg1 = (const float*)d_in[13];
  const float *ws1 = (const float*)d_in[14], *bs1 = (const float*)d_in[15];
  const float *wg2 = (const float*)d_in[16], *ag2 = (const float*)d_in[17], *bg2 = (const float*)d_in[18];
  const float *wss = (const float*)d_in[19], *bss = (const float*)d_in[20];
  const float *ws2 = (const float*)d_in[21], *bs2 = (const float*)d_in[22];
  const float *w4 = (const float*)d_in[23],  *b4  = (const float*)d_in[24];
  const float *wl = (const float*)d_in[25],  *bl  = (const float*)d_in[26];

  const int* src = eidx;
  const int* dst = eidx + E;

  char* wsb = (char*)d_ws;
  size_t off = 0;
  auto alloc = [&](size_t bytes) -> void* {
    void* p = wsb + off;
    off += (bytes + 255) & ~(size_t)255;
    return p;
  };

  // edge bf16 buffers (reused per dependency graph)
  unsigned short* e1b  = (unsigned short*)alloc((size_t)E * 128 * 2);  // e1; later xs3 scratch; later e4
  unsigned short* e2b  = (unsigned short*)alloc((size_t)E * 128 * 2);  // e2; later L5 scratch
  unsigned short* e3b  = (unsigned short*)alloc((size_t)E * 128 * 2);  // e3
  unsigned short* es1b = (unsigned short*)alloc((size_t)E * 256 * 2);  // es1; later xss2 scratch
  unsigned short* es2b = (unsigned short*)alloc((size_t)E * 256 * 2);  // es2
  unsigned short* ess1b= (unsigned short*)alloc((size_t)E * 256 * 2);  // ess1
  unsigned short* efb  = (unsigned short*)alloc((size_t)E * 64 * 2);   // edge_feat bf16
  unsigned short* xb   = (unsigned short*)alloc((size_t)N * 256 * 2);  // node staging bf16

  // packed weights
  unsigned short* wp1  = (unsigned short*)alloc((size_t)192 * 128 * 2);
  unsigned short* wp2  = (unsigned short*)alloc((size_t)384 * 128 * 2);
  unsigned short* wp3  = (unsigned short*)alloc((size_t)768 * 128 * 2);
  unsigned short* wp3p = (unsigned short*)alloc((size_t)384 * 256 * 2);
  unsigned short* wps1 = (unsigned short*)alloc((size_t)768 * 256 * 2);
  unsigned short* wpss = (unsigned short*)alloc((size_t)768 * 256 * 2);
  unsigned short* wps2 = (unsigned short*)alloc((size_t)768 * 128 * 2);
  unsigned short* wp4  = (unsigned short*)alloc((size_t)768 * 128 * 2);

  float* accA = (float*)alloc((size_t)N * 128 * 4);
  float* accB = (float*)alloc((size_t)N * 256 * 4);
  float* deg0 = (float*)alloc((size_t)N * 4);
  float* deg1 = (float*)alloc((size_t)N * 4);
  float* deg2 = (float*)alloc((size_t)N * 4);

  float* x1  = (float*)alloc((size_t)N * 128 * 4);
  float* x2  = (float*)alloc((size_t)N * 128 * 4);
  float* x3  = (float*)alloc((size_t)N * 128 * 4);
  float* x4  = (float*)alloc((size_t)N * 128 * 4);
  float* x5  = (float*)alloc((size_t)N * 128 * 4);
  float* xs3 = (float*)alloc((size_t)N * 128 * 4);
  float* x32 = (float*)alloc((size_t)N * 128 * 4);
  float* xs1 = (float*)alloc((size_t)N * 256 * 4);
  float* xp  = (float*)alloc((size_t)N * 256 * 4);
  float* xs2 = (float*)alloc((size_t)N * 256 * 4);
  float* xpp = (float*)alloc((size_t)N * 256 * 4);
  float* xss1= (float*)alloc((size_t)N * 256 * 4);
  float* xss2= (float*)alloc((size_t)N * 256 * 4);
  float* xs22= (float*)alloc((size_t)N * 256 * 4);

  float* hsb = (float*)alloc((size_t)N * 4);
  float* mxb = (float*)alloc((size_t)N * 4);
  float* denb= (float*)alloc((size_t)N * 4);
  float* numb= (float*)alloc((size_t)N * 4);
  float* scb = (float*)alloc((size_t)N * 4);
  float* nm1 = (float*)alloc((size_t)N * 4);
  float* nm2 = (float*)alloc((size_t)N * 4);
  unsigned int* ukb = (unsigned int*)alloc((size_t)N * 4);
  float* em1 = (float*)alloc((size_t)E * 4);
  float* em2 = (float*)alloc((size_t)E * 4);

  auto cdiv = [](int a, int b) { return (a + b - 1) / b; };
  auto fill = [&](float* p, float v, int n) {
    k_fill_f32<<<cdiv(n, 256), 256, 0, stream>>>(p, v, n);
  };
  auto packw = [&](const float* Wm, unsigned short* Wp, int K, int NFo) {
    k_pack_w<<<cdiv(K * NFo, 256), 256, 0, stream>>>(Wm, Wp, K, NFo);
  };
  auto pack1 = [&](const float* a, unsigned short* o, int n) {
    k_pack_bf16<<<cdiv(n, 256), 256, 0, stream>>>(a, o, n);
  };
  auto pack2 = [&](const float* a, int wa, const float* b, int wb, unsigned short* o) {
    k_pack2_bf16<<<cdiv(N * (wa + wb), 256), 256, 0, stream>>>(a, wa, b, wb, o, N);
  };
  auto elayer = [&](const unsigned short* xcat, int dx,
                    const unsigned short* eA, int wA,
                    const unsigned short* eB, int wB,
                    const unsigned short* Wp, const float* bias, int NFo,
                    unsigned short* msg, float* nacc, const float* em,
                    const float* deg, float* nodeOut) {
    int K = 2 * dx + wA + wB;
    fill(nacc, 0.f, N * NFo);
    dim3 g(E / 64, NFo / 64);  // 4 waves/block, 16 edges/wave, 64 cols/wave
    k_edge_mlp<<<g, 128, 0, stream>>>(xcat, dx, eA, wA, eB, wB, src, dst,
                                      Wp, bias, K, NFo, msg, nacc, em);
    k_node_fin<<<cdiv(N * NFo, 256), 256, 0, stream>>>(nacc, deg, nodeOut, NFo);
  };
  auto sag = [&](const float* x, const float* wg, const float* ag, const float* bg,
                 const float* emPrev, const float* nmPrev, int k,
                 float* keep, float* xOut, float* emNew, float* degNew) {
    k_dot_w<<<cdiv(N * 32, 256), 256, 0, stream>>>(x, wg, hsb, 256);
    k_gat_init<<<cdiv(N, 256), 256, 0, stream>>>(hsb, ag, mxb, denb, numb);
    k_gat_emax<<<cdiv(E, 256), 256, 0, stream>>>(hsb, src, dst, emPrev, ag, mxb);
    k_gat_esum<<<cdiv(E, 256), 256, 0, stream>>>(hsb, src, dst, emPrev, ag, mxb, denb, numb);
    k_gat_score<<<cdiv(N, 256), 256, 0, stream>>>(hsb, ag, bg, mxb, denb, numb, scb);
    k_ukey<<<cdiv(N, 256), 256, 0, stream>>>(scb, nmPrev, ukb);
    k_topk<<<1, 256, 0, stream>>>(ukb, k, keep);
    k_pool_x<<<cdiv(N * 256, 256), 256, 0, stream>>>(x, keep, scb, xOut, 256);
    k_emask<<<cdiv(E, 256), 256, 0, stream>>>(emPrev, keep, src, dst, emNew);
    fill(degNew, 0.f, N);
    k_deg<<<cdiv(E, 256), 256, 0, stream>>>(dst, emNew, degNew);
  };

  // ---- weight packing + input conversion ----
  packw(w1, wp1, 192, 128);
  packw(w2, wp2, 384, 128);
  packw(w3, wp3, 768, 128);
  packw(w3p, wp3p, 384, 256);
  packw(ws1, wps1, 768, 256);
  packw(wss, wpss, 768, 256);
  packw(ws2, wps2, 768, 128);
  packw(w4, wp4, 768, 128);
  pack1(edge_feat, efb, E * 64);

  fill(deg0, 0.f, N);
  k_deg<<<cdiv(E, 256), 256, 0, stream>>>(dst, (const float*)nullptr, deg0);

  // L1: [nf_src, nf_dst, ef] @ w1 -> x1, e1
  pack1(node_feat, xb, N * 64);
  elayer(xb, 64, efb, 64, nullptr, 0, wp1, b1, 128, e1b, accA, nullptr, deg0, x1);

  // L2: [x1_src, x1_dst, e1] @ w2 -> x2, e2
  pack1(x1, xb, N * 128);
  elayer(xb, 128, e1b, 128, nullptr, 0, wp2, b2, 128, e2b, accA, nullptr, deg0, x2);

  // L3: x=cat(x2,x1), e=cat(e2,e1) @ w3 -> x3, e3
  pack2(x2, 128, x1, 128, xb);
  elayer(xb, 256, e2b, 128, e1b, 128, wp3, b3, 128, e3b, accA, nullptr, deg0, x3);

  // L3p: [x3_src, x3_dst, e3] @ w3p -> xs1, es1
  pack1(x3, xb, N * 128);
  elayer(xb, 128, e3b, 128, nullptr, 0, wp3p, b3p, 256, es1b, accB, nullptr, deg0, xs1);

  // SAG pool 1 (k = N/2)
  sag(xs1, wg1, ag1, bg1, nullptr, nullptr, N / 2, nm1, xp, em1, deg1);

  // Ls1: [xp_src, xp_dst, es1] @ ws1 (masked em1) -> xs2, es2
  pack1(xp, xb, N * 256);
  elayer(xb, 256, es1b, 256, nullptr, 0, wps1, bs1, 256, es2b, accB, em1, deg1, xs2);

  // SAG pool 2 (k = N/4)
  sag(xs2, wg2, ag2, bg2, em1, nm1, N / 4, nm2, xpp, em2, deg2);

  // Lss #1: [xpp, es2] @ wss (masked em2) -> xss1, ess1
  pack1(xpp, xb, N * 256);
  elayer(xb, 256, es2b, 256, nullptr, 0, wpss, bss, 256, ess1b, accB, em2, deg2, xss1);

  // Lss #2: [xss1, ess1] @ wss (masked em2) -> xss2 (edge out unused -> es1b scratch)
  pack1(xss1, xb, N * 256);
  elayer(xb, 256, ess1b, 256, nullptr, 0, wpss, bss, 256, es1b, accB, em2, deg2, xss2);

  // xs2_2 = nm2 ? xs2 + xss2 : xs2
  k_resid<<<cdiv(N * 256, 256), 256, 0, stream>>>(xs2, xss2, nm2, xs22, 256);

  // Ls2: [xs2_2, es2] @ ws2 (masked em1) -> xs3 (edge out unused -> e1b scratch)
  pack1(xs22, xb, N * 256);
  elayer(xb, 256, es2b, 256, nullptr, 0, wps2, bs2, 128, e1b, accA, em1, deg1, xs3);

  // x3_2 = nm1 ? x3 + xs3 : x3
  k_resid<<<cdiv(N * 128, 256), 256, 0, stream>>>(x3, xs3, nm1, x32, 128);

  // L4: x=cat(x3_2,x2), e=cat(e3,e2) @ w4 -> x4, e4 (into e1b, now free)
  pack2(x32, 128, x2, 128, xb);
  elayer(xb, 256, e3b, 128, e2b, 128, wp4, b4, 128, e1b, accA, nullptr, deg0, x4);

  // L5: x=cat(x4,x3_2), e=cat(e4,e3) @ w4 -> x5 (edge out unused -> e2b scratch)
  pack2(x4, 128, x32, 128, xb);
  elayer(xb, 256, e1b, 128, e3b, 128, wp4, b4, 128, e2b, accA, nullptr, deg0, x5);

  // out = x5 @ wl + bl
  k_final<<<cdiv(N, 256), 256, 0, stream>>>(x5, wl, bl, (float*)d_out);
}
